// Attention_base_25709674234651
// MI455X (gfx1250) — matrix operand order repaired and checked by reading, not compiled, not run on hardware
//
#include <hip/hip_runtime.h>
#include <hip/hip_bf16.h>

// ---------------------------------------------------------------------------
// Attention block for MI455X (gfx1250, wave32, WMMA).
//   qkv = x @ w_qkv + b_qkv  ->  per-head flash attention  ->  out [B,N,C] f32
// f16 storage, f32 WMMA accumulation (v_wmma_f32_16x16x32_f16).
// Workspace (f16): Q[B,H,N,D] (pre-scaled by 1/sqrt(D)), K[B,H,N,D],
//                  Vt[B,H,D,N]  -> 24 MB of d_ws.
// K/V attention staging uses CDNA5 async global->LDS copies
// (global_load_async_to_lds_b128, ASYNCcnt + s_wait_asynccnt).
// ---------------------------------------------------------------------------

#define DIM_C 1024
#define SEQ   2048
#define NBATCH 2
#define NH    16
#define HD    64
#define NOUT  3072   // 3*DIM_C

typedef __attribute__((ext_vector_type(16))) _Float16 v16h;
typedef __attribute__((ext_vector_type(8)))  _Float16 v8h;
typedef __attribute__((ext_vector_type(4)))  _Float16 v4h;
typedef __attribute__((ext_vector_type(8)))  float    v8f;
typedef __attribute__((ext_vector_type(4)))  int      v4i;

// address-space-qualified b128 beat pointers for the async builtin
typedef v4i __attribute__((address_space(1))) as1_v4i;
typedef v4i __attribute__((address_space(3))) as3_v4i;

static __device__ __forceinline__ v8f wmma_f16(v16h a, v16h b, v8f c) {
  // (neg_a, A, neg_b, B, c_mod, C, reuse_a, reuse_b)
  return __builtin_amdgcn_wmma_f32_16x16x32_f16(false, a, false, b, (short)0, c,
                                                false, false);
}

// ---- CDNA5 async global->LDS copy (16 bytes per lane), guarded ------------
#if __has_builtin(__builtin_amdgcn_global_load_async_to_lds_b128)
#define HAVE_ASYNC_LDS 1
#else
#define HAVE_ASYNC_LDS 0
#endif

static __device__ __forceinline__ void stage16(const _Float16* src,
                                               _Float16* dst) {
#if HAVE_ASYNC_LDS
  __builtin_amdgcn_global_load_async_to_lds_b128(
      (as1_v4i*)(_Float16*)src, (as3_v4i*)dst, 0, 0);
#else
  *(v8h*)dst = *(const v8h*)src;
#endif
}

static __device__ __forceinline__ void stage_wait() {
#if HAVE_ASYNC_LDS
#if __has_builtin(__builtin_amdgcn_s_wait_asynccnt)
  __builtin_amdgcn_s_wait_asynccnt(0);
#else
  asm volatile("s_wait_asynccnt 0x0" ::: "memory");
#endif
#endif
}

// ---------------------------------------------------------------------------
// Kernel 1: qkv = x @ w + b.  M=4096 (B*N), K=1024, N=3072.
// Block = 128 threads (4 waves), 64x128 tile; each wave owns 32x64 = 2x4
// WMMA tiles (8 wmma per k-step). W panel is stored in LDS pre-transposed
// ([col][k]) so B fragments are contiguous 32-byte reads. (Staging here must
// pass through VGPRs anyway for the f32->f16 conversion, so no async path.)
// ---------------------------------------------------------------------------
__global__ __launch_bounds__(128) void qkv_gemm_kernel(
    const float* __restrict__ x, const float* __restrict__ w,
    const float* __restrict__ bias,
    _Float16* __restrict__ qout, _Float16* __restrict__ kout,
    _Float16* __restrict__ vtout)
{
  __shared__ __align__(32) _Float16 As[64 * 32];    // [row][k]
  __shared__ __align__(32) _Float16 BsT[128 * 32];  // [col][k]  (fragment layout)

  const int t    = threadIdx.x;
  const int lane = t & 31;
  const int wv   = t >> 5;   // wave 0..3
  const int wm   = wv >> 1;  // 32-row block
  const int wn   = wv & 1;   // 64-col block
  const int hi   = lane >> 4;
  const int ln   = lane & 15;

  const int m0 = blockIdx.x * 64;    // 4096/64  = 64
  const int n0 = blockIdx.y * 128;   // 3072/128 = 24

  // B staging: 4k x 8col patch per thread
  const int ck = t & 7;    // k chunk (4 rows)
  const int cc = t >> 3;   // col chunk (8 cols), 0..15

  v8f acc[2][4] = {};

  for (int kk = 0; kk < DIM_C; kk += 32) {
    // ---- stage A panel: x[m0..+64][kk..+32] -> f16 LDS [row][k] ----
    {
      const int r = t >> 1;
      const int c = (t & 1) * 16;
      const float* src = x + (size_t)(m0 + r) * DIM_C + kk + c;
      v8h h0, h1;
      #pragma unroll
      for (int i = 0; i < 8; i += 4) {
        float4 f = *(const float4*)(src + i);
        h0[i + 0] = (_Float16)f.x; h0[i + 1] = (_Float16)f.y;
        h0[i + 2] = (_Float16)f.z; h0[i + 3] = (_Float16)f.w;
      }
      #pragma unroll
      for (int i = 0; i < 8; i += 4) {
        float4 f = *(const float4*)(src + 8 + i);
        h1[i + 0] = (_Float16)f.x; h1[i + 1] = (_Float16)f.y;
        h1[i + 2] = (_Float16)f.z; h1[i + 3] = (_Float16)f.w;
      }
      *(v8h*)&As[r * 32 + c]     = h0;
      *(v8h*)&As[r * 32 + c + 8] = h1;
    }
    // ---- stage W panel transposed: w[kk..+32][n0..+128] -> LDS [col][k] ----
    {
      _Float16 tmp[4][8];
      #pragma unroll
      for (int kr = 0; kr < 4; ++kr) {
        const float* src = w + (size_t)(kk + ck * 4 + kr) * NOUT + n0 + cc * 8;
        float4 f0 = *(const float4*)(src);
        float4 f1 = *(const float4*)(src + 4);
        tmp[kr][0] = (_Float16)f0.x; tmp[kr][1] = (_Float16)f0.y;
        tmp[kr][2] = (_Float16)f0.z; tmp[kr][3] = (_Float16)f0.w;
        tmp[kr][4] = (_Float16)f1.x; tmp[kr][5] = (_Float16)f1.y;
        tmp[kr][6] = (_Float16)f1.z; tmp[kr][7] = (_Float16)f1.w;
      }
      #pragma unroll
      for (int i = 0; i < 8; ++i) {
        v4h pk;
        pk[0] = tmp[0][i]; pk[1] = tmp[1][i];
        pk[2] = tmp[2][i]; pk[3] = tmp[3][i];
        *(v4h*)&BsT[(cc * 8 + i) * 32 + ck * 4] = pk;  // b64 store
      }
    }
    // prefetch next panels while this k-step computes (global_prefetch_b8)
    if (kk + 32 < DIM_C) {
      __builtin_prefetch(x + (size_t)(m0 + (t >> 1)) * DIM_C + kk + 32, 0, 1);
      __builtin_prefetch(w + (size_t)(kk + 32 + ck * 4) * NOUT + n0 + cc * 8, 0, 1);
    }
    __syncthreads();

    // A fragments: lane = M row; elem e -> k = hi*16 + e
    v16h afrag[2];
    #pragma unroll
    for (int s = 0; s < 2; ++s) {
      const _Float16* base = &As[(wm * 32 + s * 16 + ln) * 32 + hi * 16];
      v8h lo = *(const v8h*)(base);
      v8h h8 = *(const v8h*)(base + 8);
      #pragma unroll
      for (int e = 0; e < 8; ++e) { afrag[s][e] = lo[e]; afrag[s][8 + e] = h8[e]; }
    }
    // B fragments: lane = col; elem e -> k = hi*16 + e  (contiguous 32B)
    #pragma unroll
    for (int j = 0; j < 4; ++j) {
      const v16h bf = *(const v16h*)&BsT[(wn * 64 + j * 16 + ln) * 32 + hi * 16];
      #pragma unroll
      for (int i = 0; i < 2; ++i)
        acc[i][j] = wmma_f16(afrag[i], bf, acc[i][j]);
    }
    __syncthreads();
  }

  // ---- epilogue: bias + scatter into Q (scaled), K, Vt ----
  #pragma unroll
  for (int j = 0; j < 4; ++j) {
    const int c      = n0 + wn * 64 + j * 16 + ln;   // column in [0,3072)
    const float bb   = bias[c];
    const int part   = c >> 10;        // 0=q 1=k 2=v (uniform per block)
    const int within = c & 1023;
    const int h      = within >> 6;
    const int d      = within & 63;
    #pragma unroll
    for (int i = 0; i < 2; ++i) {
      const int mbase = m0 + wm * 32 + i * 16 + hi * 8;  // multiple of 8
      const int b     = mbase >> 11;
      const int nb    = mbase & 2047;
      if (part == 2) {
        // Vt[b,h,d,n]: 8 consecutive n -> one b128 store
        v8h vv;
        #pragma unroll
        for (int r = 0; r < 8; ++r) vv[r] = (_Float16)(acc[i][j][r] + bb);
        *(v8h*)&vtout[((size_t)((b * NH + h) * HD) + d) * SEQ + nb] = vv;
      } else {
        #pragma unroll
        for (int r = 0; r < 8; ++r) {
          const float val = acc[i][j][r] + bb;
          const size_t idx = ((size_t)((b * NH + h) * SEQ) + nb + r) * HD + d;
          if (part == 0) qout[idx] = (_Float16)(val * 0.125f);  // 64^-0.5
          else           kout[idx] = (_Float16)val;
        }
      }
    }
  }
}

// ---------------------------------------------------------------------------
// Kernel 2: flash attention. Block = 128 threads (4 waves) = (b, h, 64 q
// rows); each wave owns 16 q rows. 64-key tiles; K/V staged once per block
// via async global->LDS copies. All B fragments are contiguous 32-byte LDS
// reads. Per tile per wave: 8 WMMA (S) + online softmax + 8 WMMA (O += P V).
// ---------------------------------------------------------------------------
__global__ __launch_bounds__(128) void attn_kernel(
    const _Float16* __restrict__ q, const _Float16* __restrict__ k,
    const _Float16* __restrict__ vt, float* __restrict__ out)
{
  __shared__ __align__(32) _Float16 Ks[64 * 64];      // [key][d]
  __shared__ __align__(32) _Float16 Vs[64 * 64];      // [d][key]
  __shared__ __align__(32) _Float16 Ps[4][16 * 64];   // per-wave [qrow][key]

  const int t    = threadIdx.x;
  const int lane = t & 31;
  const int wv   = t >> 5;
  const int hi   = lane >> 4;
  const int ln   = lane & 15;

  const int bh = blockIdx.y;                // b*NH + h
  const int q0 = blockIdx.x * 64 + wv * 16; // this wave's first query row

  // ---- persistent Q A-fragments (2 k-steps over D=64) ----
  const _Float16* qbase = q + ((size_t)bh * SEQ + q0) * HD;
  v16h qf[2];
  #pragma unroll
  for (int ks = 0; ks < 2; ++ks) {
    const _Float16* p = qbase + (size_t)ln * HD + ks * 32 + hi * 16;
    v8h lo = *(const v8h*)(p);
    v8h h8 = *(const v8h*)(p + 8);
    #pragma unroll
    for (int e = 0; e < 8; ++e) { qf[ks][e] = lo[e]; qf[ks][8 + e] = h8[e]; }
  }

  float mrow[8], lrow[8];
  #pragma unroll
  for (int r = 0; r < 8; ++r) { mrow[r] = -1e30f; lrow[r] = 0.0f; }
  v8f o[4] = {};

  for (int k0 = 0; k0 < SEQ; k0 += 64) {
    __syncthreads();  // previous tile's fragments fully consumed
    // ---- stage K tile [64 key][64 d]: contiguous 8 KB, async to LDS ----
    {
      const _Float16* src = k + ((size_t)bh * SEQ + k0) * HD + t * 32;
      _Float16* dst = &Ks[t * 32];
      #pragma unroll
      for (int i = 0; i < 32; i += 8) stage16(src + i, dst + i);
    }
    // ---- stage Vt tile [64 d][64 key]: row-wise, async to LDS ----
    {
      const int d = t >> 1;
      const int c = (t & 1) * 32;
      const _Float16* src = vt + ((size_t)bh * HD + d) * SEQ + k0 + c;
      _Float16* dst = &Vs[d * 64 + c];
      #pragma unroll
      for (int i = 0; i < 32; i += 8) stage16(src + i, dst + i);
    }
    // prefetch next tiles (global_prefetch_b8)
    if (k0 + 64 < SEQ) {
      __builtin_prefetch(k + ((size_t)bh * SEQ + k0 + 64) * HD + t * 32, 0, 1);
      __builtin_prefetch(vt + ((size_t)bh * HD + (t >> 1)) * SEQ + k0 + 64 + (t & 1) * 32, 0, 1);
    }
    stage_wait();     // this wave's async copies landed in LDS
    __syncthreads();  // all waves' copies landed

    // ---- S = Q Kt : 16 x 64 per wave; B frag = contiguous 32B of Ks ----
    v8f s[4] = {};
    #pragma unroll
    for (int cb = 0; cb < 4; ++cb) {
      #pragma unroll
      for (int ks = 0; ks < 2; ++ks) {
        const v16h bf = *(const v16h*)&Ks[(cb * 16 + ln) * 64 + ks * 32 + hi * 16];
        s[cb] = wmma_f16(qf[ks], bf, s[cb]);
      }
    }

    // ---- online softmax (rows live in 16-lane halves; offsets<16 stay put)
    float mnew[8];
    #pragma unroll
    for (int r = 0; r < 8; ++r) mnew[r] = mrow[r];
    #pragma unroll
    for (int cb = 0; cb < 4; ++cb)
      #pragma unroll
      for (int r = 0; r < 8; ++r) mnew[r] = fmaxf(mnew[r], s[cb][r]);
    #pragma unroll
    for (int off = 1; off < 16; off <<= 1)
      #pragma unroll
      for (int r = 0; r < 8; ++r)
        mnew[r] = fmaxf(mnew[r], __shfl_xor(mnew[r], off, 32));

    float rs[8];
    #pragma unroll
    for (int r = 0; r < 8; ++r) rs[r] = 0.0f;
    #pragma unroll
    for (int cb = 0; cb < 4; ++cb) {
      #pragma unroll
      for (int r = 0; r < 8; ++r) {
        const float pv = __expf(s[cb][r] - mnew[r]);
        rs[r] += pv;
        // C layout -> LDS [qrow][key]  (the one unavoidable scatter)
        Ps[wv][(r + hi * 8) * 64 + cb * 16 + ln] = (_Float16)pv;
      }
    }
    #pragma unroll
    for (int off = 1; off < 16; off <<= 1)
      #pragma unroll
      for (int r = 0; r < 8; ++r)
        rs[r] += __shfl_xor(rs[r], off, 32);

    #pragma unroll
    for (int r = 0; r < 8; ++r) {
      const float al = __expf(mrow[r] - mnew[r]);
      lrow[r] = lrow[r] * al + rs[r];
      mrow[r] = mnew[r];
      #pragma unroll
      for (int cb = 0; cb < 4; ++cb) o[cb][r] *= al;
    }

    // make the wave-private P stores visible to our own ds loads
    asm volatile("s_wait_dscnt 0" ::: "memory");

    // ---- O += P V : A frag from Ps, B frag = contiguous 32B of Vs ----
    v16h pf[2];
    #pragma unroll
    for (int ks = 0; ks < 2; ++ks) {
      const _Float16* p = &Ps[wv][ln * 64 + ks * 32 + hi * 16];
      v8h lo = *(const v8h*)(p);
      v8h h8 = *(const v8h*)(p + 8);
      #pragma unroll
      for (int e = 0; e < 8; ++e) { pf[ks][e] = lo[e]; pf[ks][8 + e] = h8[e]; }
    }
    #pragma unroll
    for (int cb = 0; cb < 4; ++cb) {
      #pragma unroll
      for (int ks = 0; ks < 2; ++ks) {
        const v16h bv = *(const v16h*)&Vs[(cb * 16 + ln) * 64 + ks * 32 + hi * 16];
        o[cb] = wmma_f16(pf[ks], bv, o[cb]);
      }
    }
  }

  // ---- normalize and write out [B,N,C] f32 ----
  const int b = bh >> 4;
  const int h = bh & 15;
  #pragma unroll
  for (int cb = 0; cb < 4; ++cb) {
    const int d = cb * 16 + ln;
    #pragma unroll
    for (int r = 0; r < 8; ++r) {
      const int qr = q0 + r + hi * 8;
      out[((size_t)(b * SEQ + qr)) * DIM_C + h * HD + d] = o[cb][r] / lrow[r];
    }
  }
}

extern "C" void kernel_launch(void* const* d_in, const int* in_sizes, int n_in,
                              void* d_out, int out_size, void* d_ws, size_t ws_size,
                              hipStream_t stream) {
  (void)in_sizes; (void)n_in; (void)out_size; (void)ws_size;
  const float* x    = (const float*)d_in[0];
  const float* w    = (const float*)d_in[1];
  const float* bias = (const float*)d_in[2];
  float* out        = (float*)d_out;

  // f16 workspace: Q | K | Vt, each B*H*SEQ*HD halves (8 MB) -> 24 MB total.
  _Float16* wsh = (_Float16*)d_ws;
  const size_t QSZ = (size_t)NBATCH * NH * SEQ * HD;
  _Float16* qws  = wsh;
  _Float16* kws  = wsh + QSZ;
  _Float16* vtws = wsh + 2 * QSZ;

  dim3 g1(4096 / 64, NOUT / 128), b1(128);
  qkv_gemm_kernel<<<g1, b1, 0, stream>>>(x, w, bias, qws, kws, vtws);

  dim3 g2(SEQ / 64, NBATCH * NH), b2(128);
  attn_kernel<<<g2, b2, 0, stream>>>(qws, kws, vtws, out);
}
